// SharedScaleLinear_13718125543810
// MI455X (gfx1250) — matrix operand order repaired and checked by reading, not compiled, not run on hardware
//
#include <hip/hip_runtime.h>

#define TOKENS 8192
#define IN_F   3072
#define OUT_F  8192
#define GROUP  128
#define NG     (IN_F / GROUP)   // 24 scale groups per row

#define BM 128
#define BN 128
#define BK 128
#define PAD 8
#define LDK (BK + PAD)          // 136 halves -> 272B row stride (4-bank rotate)
#define KITERS (IN_F / BK)      // 24

typedef __attribute__((ext_vector_type(16))) _Float16 v16h;
typedef __attribute__((ext_vector_type(8)))  _Float16 v8h;
typedef __attribute__((ext_vector_type(4)))  _Float16 v4h;
typedef __attribute__((ext_vector_type(8)))  float    v8f;
typedef int vs4i __attribute__((vector_size(16)));   // matches builtin param type

#define AS1 __attribute__((address_space(1)))
#define AS3 __attribute__((address_space(3)))

#if __has_builtin(__builtin_amdgcn_global_load_async_to_lds_b128)
#define USE_ASYNC_LDS 1
#else
#define USE_ASYNC_LDS 0
#endif

// 16-byte global->LDS copy: async (no VGPR data movement, ASYNCcnt) when the
// toolchain exposes the gfx1250 builtin, else a plain b128 load/store.
__device__ __forceinline__ void copy16(const void* g, void* l) {
#if USE_ASYNC_LDS
    __builtin_amdgcn_global_load_async_to_lds_b128(
        (AS1 vs4i*)(uintptr_t)g,
        (AS3 vs4i*)(unsigned)(uintptr_t)l,   // low 32 bits of generic addr = LDS offset
        0, 0);
#else
    *(v8h*)l = *(const v8h*)g;
#endif
}

__device__ __forceinline__ void wait_copies() {
#if USE_ASYNC_LDS
#if __has_builtin(__builtin_amdgcn_s_wait_asynccnt)
    __builtin_amdgcn_s_wait_asynccnt(0);
#else
    asm volatile("s_wait_asynccnt 0x0" ::: "memory");
#endif
#endif
}

// ---------------------------------------------------------------------------
// Prepass 1: dequantize int4 weights (1 byte per int32) -> f16, once.
// Same byte count as the packed form, so GEMM bandwidth is unchanged while all
// per-tile dequant VALU (redone 64x in the fused kernel) disappears.
// ---------------------------------------------------------------------------
__global__ __launch_bounds__(256)
void dequant_w_kernel(const int* __restrict__ wp, const float* __restrict__ sp,
                      _Float16* __restrict__ wh)
{
    const int idx = blockIdx.x * 256 + threadIdx.x;     // over OUT_F * IN_F/2
    if (idx >= OUT_F * (IN_F / 2)) return;
    const int row = idx / (IN_F / 2);
    const int c   = idx - row * (IN_F / 2);
    const unsigned b = (unsigned)wp[idx] & 0xffu;
    int lo = (int)(b & 15u);        lo -= (lo & 8) << 1;
    int hi = (int)((b >> 4) & 15u); hi -= (hi & 8) << 1;
    const float sc = sp[(size_t)row * NG + (c * 2) / GROUP];
    union { _Float16 h[2]; unsigned u; } p;
    p.h[0] = (_Float16)((float)lo * sc);
    p.h[1] = (_Float16)((float)hi * sc);
    *(unsigned*)(wh + (size_t)row * IN_F + c * 2) = p.u;  // coalesced 4B store
}

// ---------------------------------------------------------------------------
// Prepass 2: x f32 -> f16, once (instead of once per N-tile = 64x).
// ---------------------------------------------------------------------------
__global__ __launch_bounds__(256)
void cvt_x_kernel(const float* __restrict__ xg, _Float16* __restrict__ xh)
{
    const size_t idx = (size_t)blockIdx.x * 256 + threadIdx.x;   // float4 units
    if (idx >= (size_t)TOKENS * IN_F / 4) return;
    const float4 v = ((const float4*)xg)[idx];
    v4h h;
    h[0] = (_Float16)v.x; h[1] = (_Float16)v.y;
    h[2] = (_Float16)v.z; h[3] = (_Float16)v.w;
    ((v4h*)xh)[idx] = h;
}

// ---------------------------------------------------------------------------
// Main GEMM: f16 x f16 -> f32, double-buffered LDS, async global->LDS staging
// overlapped with v_wmma_f32_16x16x32_f16 compute.
// ---------------------------------------------------------------------------
__global__ __launch_bounds__(256)
void gemm_f16_wmma(const _Float16* __restrict__ xh,
                   const _Float16* __restrict__ wh,
                   const float* __restrict__ bp,
                   float* __restrict__ out)
{
    __shared__ _Float16 xsm[2][BM][LDK];   // 2 x 34816 B
    __shared__ _Float16 wsm[2][BN][LDK];   // 2 x 34816 B  (total 139264 B)

    const int tid   = threadIdx.x;
    const int lane  = tid & 31;
    const int wave  = tid >> 5;
    const int bm    = blockIdx.y * BM;
    const int bn    = blockIdx.x * BN;

    const int mrow  = (wave & 3) * 32;     // wave sub-tile: 32(M) x 64(N)
    const int ncol  = (wave >> 2) * 64;
    const int lrow  = lane & 15;
    const int lhalf = lane >> 4;

    v8f acc[2][4];
#pragma unroll
    for (int mi = 0; mi < 2; ++mi)
#pragma unroll
        for (int ni = 0; ni < 4; ++ni)
            acc[mi][ni] = (v8f)0.0f;

    // stage one 128x128 f16 tile of x and of w into LDS buffer `buf`
    auto stage = [&](int it, int buf) {
        const int k0 = it * BK;
#pragma unroll
        for (int j = 0; j < 8; ++j) {
            const int idx = j * 256 + tid;       // 2048 16B-chunks per array
            const int row = idx >> 4;            // 16 chunks per 256B row
            const int ch  = idx & 15;
            copy16(xh + (size_t)(bm + row) * IN_F + k0 + ch * 8,
                   &xsm[buf][row][ch * 8]);
            copy16(wh + (size_t)(bn + row) * IN_F + k0 + ch * 8,
                   &wsm[buf][row][ch * 8]);
        }
    };

    stage(0, 0);
    wait_copies();
    __syncthreads();

    for (int it = 0; it < KITERS; ++it) {
        const int buf = it & 1;
        if (it + 1 < KITERS) stage(it + 1, buf ^ 1);   // overlap with compute

#pragma unroll
        for (int ks = 0; ks < BK / 32; ++ks) {
            v16h a[2], bmat[4];
#pragma unroll
            for (int mi = 0; mi < 2; ++mi) {
                // A 16x32 f16: lane<16 -> K[0..8)+[16..24); lane>=16 -> +8
                const _Float16* ap = &xsm[buf][mrow + mi * 16 + lrow][ks * 32 + lhalf * 8];
                const v8h a0 = *(const v8h*)ap;
                const v8h a1 = *(const v8h*)(ap + 16);
                a[mi] = __builtin_shufflevector(a0, a1,
                        0,1,2,3,4,5,6,7,8,9,10,11,12,13,14,15);
            }
#pragma unroll
            for (int ni = 0; ni < 4; ++ni) {
                // B 32x16 f16: lane<16 -> N=lane, K[0..8)+[16..24); lane>=16 -> +8
                const _Float16* bptr = &wsm[buf][ncol + ni * 16 + lrow][ks * 32 + lhalf * 8];
                const v8h b0 = *(const v8h*)bptr;
                const v8h b1 = *(const v8h*)(bptr + 16);
                bmat[ni] = __builtin_shufflevector(b0, b1,
                        0,1,2,3,4,5,6,7,8,9,10,11,12,13,14,15);
            }
#pragma unroll
            for (int mi = 0; mi < 2; ++mi)
#pragma unroll
                for (int ni = 0; ni < 4; ++ni)
                    acc[mi][ni] = __builtin_amdgcn_wmma_f32_16x16x32_f16(
                        false, a[mi], false, bmat[ni],
                        (short)0, acc[mi][ni], false, false);
        }

        wait_copies();      // next buffer's async copies done
        __syncthreads();    // visible to all waves; also fences fallback ds stores
    }

    // epilogue: bias add + f32 store, per C/D layout (lane=N, VGPR j = M row)
    const int rsel = lhalf * 8;
#pragma unroll
    for (int ni = 0; ni < 4; ++ni) {
        const int gcol = bn + ncol + ni * 16 + lrow;
        const float bv = bp[gcol];
#pragma unroll
        for (int mi = 0; mi < 2; ++mi) {
            const int rbase = bm + mrow + mi * 16 + rsel;
#pragma unroll
            for (int j = 0; j < 8; ++j)
                out[(size_t)(rbase + j) * OUT_F + gcol] = acc[mi][ni][j] + bv;
        }
    }
}

// ---------------------------------------------------------------------------
// Fallback: fully fused kernel (round-1 design) if workspace is too small.
// ---------------------------------------------------------------------------
__global__ __launch_bounds__(256)
void w4a16_fused_gemm(const float* __restrict__ xg,
                      const int*   __restrict__ wp,
                      const float* __restrict__ sp,
                      const float* __restrict__ bp,
                      float*       __restrict__ out)
{
    __shared__ _Float16 xsm[BM][LDK];
    __shared__ _Float16 wsm[BN][LDK];

    const int tid   = threadIdx.x;
    const int lane  = tid & 31;
    const int wave  = tid >> 5;
    const int bm    = blockIdx.y * BM;
    const int bn    = blockIdx.x * BN;
    const int mrow  = (wave & 3) * 32;
    const int ncol  = (wave >> 2) * 64;
    const int lrow  = lane & 15;
    const int lhalf = lane >> 4;

    v8f acc[2][4];
#pragma unroll
    for (int mi = 0; mi < 2; ++mi)
#pragma unroll
        for (int ni = 0; ni < 4; ++ni)
            acc[mi][ni] = (v8f)0.0f;

    for (int it = 0; it < KITERS; ++it) {
        const int k0 = it * BK;
#pragma unroll
        for (int j = 0; j < 16; ++j) {
            const int idx = j * 256 + tid;
            const int row = idx >> 5;
            const int c4  = idx & 31;
            const float4 v = *(const float4*)(xg + (size_t)(bm + row) * IN_F + k0 + c4 * 4);
            v4h h;
            h[0] = (_Float16)v.x; h[1] = (_Float16)v.y;
            h[2] = (_Float16)v.z; h[3] = (_Float16)v.w;
            *(v4h*)(&xsm[row][c4 * 4]) = h;
        }
        const int kg = k0 / GROUP;
#pragma unroll
        for (int j = 0; j < 32; ++j) {
            const int idx = j * 256 + tid;
            const int row = idx >> 6;
            const int c   = idx & 63;
            const unsigned pv = (unsigned)wp[(size_t)(bn + row) * (IN_F / 2) + (k0 >> 1) + c];
            const unsigned b  = pv & 0xffu;
            int lo = (int)(b & 15u);        lo -= (lo & 8) << 1;
            int hi = (int)((b >> 4) & 15u); hi -= (hi & 8) << 1;
            const float sc = sp[(size_t)(bn + row) * NG + kg];
            union { _Float16 h[2]; unsigned u; } p;
            p.h[0] = (_Float16)((float)lo * sc);
            p.h[1] = (_Float16)((float)hi * sc);
            *(unsigned*)(&wsm[row][c * 2]) = p.u;
        }
        __syncthreads();
#pragma unroll
        for (int ks = 0; ks < BK / 32; ++ks) {
            v16h a[2], bmat[4];
#pragma unroll
            for (int mi = 0; mi < 2; ++mi) {
                const _Float16* ap = &xsm[mrow + mi * 16 + lrow][ks * 32 + lhalf * 8];
                const v8h a0 = *(const v8h*)ap;
                const v8h a1 = *(const v8h*)(ap + 16);
                a[mi] = __builtin_shufflevector(a0, a1,
                        0,1,2,3,4,5,6,7,8,9,10,11,12,13,14,15);
            }
#pragma unroll
            for (int ni = 0; ni < 4; ++ni) {
                const _Float16* bptr = &wsm[ncol + ni * 16 + lrow][ks * 32 + lhalf * 8];
                const v8h b0 = *(const v8h*)bptr;
                const v8h b1 = *(const v8h*)(bptr + 16);
                bmat[ni] = __builtin_shufflevector(b0, b1,
                        0,1,2,3,4,5,6,7,8,9,10,11,12,13,14,15);
            }
#pragma unroll
            for (int mi = 0; mi < 2; ++mi)
#pragma unroll
                for (int ni = 0; ni < 4; ++ni)
                    acc[mi][ni] = __builtin_amdgcn_wmma_f32_16x16x32_f16(
                        false, a[mi], false, bmat[ni],
                        (short)0, acc[mi][ni], false, false);
        }
        __syncthreads();
    }

    const int rsel = lhalf * 8;
#pragma unroll
    for (int ni = 0; ni < 4; ++ni) {
        const int gcol = bn + ncol + ni * 16 + lrow;
        const float bv = bp[gcol];
#pragma unroll
        for (int mi = 0; mi < 2; ++mi) {
            const int rbase = bm + mrow + mi * 16 + rsel;
#pragma unroll
            for (int j = 0; j < 8; ++j)
                out[(size_t)(rbase + j) * OUT_F + gcol] = acc[mi][ni][j] + bv;
        }
    }
}

extern "C" void kernel_launch(void* const* d_in, const int* in_sizes, int n_in,
                              void* d_out, int out_size, void* d_ws, size_t ws_size,
                              hipStream_t stream) {
    const float* x  = (const float*)d_in[0];   // [8192, 3072] f32
    const int*   w  = (const int*)d_in[1];     // [8192, 1536] int32 (uint8 payload)
    const float* s  = (const float*)d_in[2];   // [8192, 24] f32
    const float* b  = (const float*)d_in[3];   // [8192] f32
    float* out = (float*)d_out;                // [8192, 8192] f32

    const size_t WH_BYTES = (size_t)OUT_F * IN_F * sizeof(_Float16);  // 48 MB
    const size_t XH_BYTES = (size_t)TOKENS * IN_F * sizeof(_Float16); // 48 MB
    dim3 grid(OUT_F / BN, TOKENS / BM);        // (64, 64)

    if (ws_size >= WH_BYTES + XH_BYTES) {
        _Float16* wh = (_Float16*)d_ws;
        _Float16* xh = (_Float16*)((char*)d_ws + WH_BYTES);
        dequant_w_kernel<<<OUT_F * (IN_F / 2) / 256, 256, 0, stream>>>(w, s, wh);
        cvt_x_kernel<<<(int)((size_t)TOKENS * IN_F / 4 / 256), 256, 0, stream>>>(x, xh);
        gemm_f16_wmma<<<grid, dim3(256), 0, stream>>>(xh, wh, b, out);
    } else {
        w4a16_fused_gemm<<<grid, dim3(256), 0, stream>>>(x, w, s, b, out);
    }
}
